// RecurrenceAttention_120259085155
// MI455X (gfx1250) — matrix operand order repaired and checked by reading, not compiled, not run on hardware
//
#include <hip/hip_runtime.h>

#define D_MODEL 1024
#define D_INNER 1024
#define N_HEADS 16
#define D_HEAD  64
#define BATCH   2
#define CUR     1024
#define PREV    1024
#define TOT     2048

typedef __attribute__((ext_vector_type(16))) __bf16 v16bf;
typedef __attribute__((ext_vector_type(8)))  float  f32x8;
typedef __attribute__((ext_vector_type(4)))  unsigned int u32x4;
typedef __attribute__((ext_vector_type(8)))  int i32x8;
typedef __attribute__((ext_vector_type(4)))  int i32x4;

#if defined(__has_builtin)
#  if __has_builtin(__builtin_amdgcn_tensor_load_to_lds)
#    define HAS_TDM 1
#  endif
#endif
#ifndef HAS_TDM
#  define HAS_TDM 0
#endif

struct U128x2 { uint4 lo, hi; };

__device__ __forceinline__ unsigned short f2bf(float f) {
  unsigned u = __builtin_bit_cast(unsigned, f);
  unsigned r = u + 0x7FFFu + ((u >> 16) & 1u);
  return (unsigned short)(r >> 16);
}
__device__ __forceinline__ float bf2f(unsigned short h) {
  unsigned u = ((unsigned)h) << 16;
  return __builtin_bit_cast(float, u);
}

// A-fragment (16x32 bf16, M x K): lane l<16 -> row l, K {0..7,16..23}; lane>=16 -> row l-16, K {8..15,24..31}
__device__ __forceinline__ v16bf load_a_frag(const unsigned short* base, int ld) {
  const int lane = threadIdx.x & 31;
  const int row  = lane & 15;
  const int cb   = (lane >> 4) << 3;
  U128x2 t;
  t.lo = *reinterpret_cast<const uint4*>(base + row * ld + cb);
  t.hi = *reinterpret_cast<const uint4*>(base + row * ld + cb + 16);
  return __builtin_bit_cast(v16bf, t);
}
// B-fragment (32x16 bf16, K x N) from B^T stored row-major (N rows of length K)
__device__ __forceinline__ v16bf load_b_frag(const unsigned short* base, int ld) {
  const int lane = threadIdx.x & 31;
  const int n  = lane & 15;
  const int kb = (lane >> 4) << 3;
  U128x2 t;
  t.lo = *reinterpret_cast<const uint4*>(base + n * ld + kb);
  t.hi = *reinterpret_cast<const uint4*>(base + n * ld + kb + 16);
  return __builtin_bit_cast(v16bf, t);
}
__device__ __forceinline__ f32x8 wmma_bf16(v16bf a, v16bf b, f32x8 c) {
  return __builtin_amdgcn_wmma_f32_16x16x32_bf16(false, a, false, b, (short)0, c, false, false);
}

// CDNA5 async global->LDS (ASYNCcnt tracked); per-lane 16B copy, no VGPR round trip
__device__ __forceinline__ void async_load_b128(unsigned lds_off, const void* gaddr) {
  asm volatile("global_load_async_to_lds_b128 %0, %1, off"
               :: "v"(lds_off), "v"((unsigned long long)(uintptr_t)gaddr)
               : "memory");
}

#if HAS_TDM
// Tensor Data Mover: 2D tile (rows x 64 bf16 elements) from strided matrix -> LDS.
// D# layout per CDNA5 ISA ch.8 (group0: ctrl/lds/global addr; group1: dims/strides).
// This toolchain's builtin takes 6 args: (g0, g1, g2, g3, extra, cpol).
__device__ __forceinline__ void tdm_load_tile_2d(unsigned lds_off, const void* gaddr,
                                                 unsigned tile_w, unsigned tile_h,
                                                 unsigned long long stride_elems) {
  const unsigned long long ga = (unsigned long long)(uintptr_t)gaddr;
  u32x4 g0;
  g0[0] = 1u;                                            // count=1, user mode
  g0[1] = lds_off;                                       // lds_addr (bytes)
  g0[2] = (unsigned)(ga & 0xFFFFFFFFu);                  // global_addr[31:0]
  g0[3] = (unsigned)((ga >> 32) & 0x01FFFFFFu) | (2u << 30); // addr[56:32] | type=2
  i32x8 g1;
  g1[0] = (int)(1u << 16);                               // data_size=1 -> 2 bytes
  g1[1] = (int)((tile_w & 0xFFFFu) << 16);               // tensor_dim0[15:0]
  g1[2] = (int)((tile_w >> 16) | ((tile_h & 0xFFFFu) << 16)); // dim0 hi | dim1 lo
  g1[3] = (int)((tile_h >> 16) | (tile_w << 16));        // dim1 hi | tile_dim0
  g1[4] = (int)(tile_h & 0xFFFFu);                       // tile_dim1 (tile_dim2=0)
  g1[5] = (int)(stride_elems & 0xFFFFFFFFu);             // tensor_dim0_stride[31:0]
  g1[6] = (int)((stride_elems >> 32) & 0xFFFFu);         // stride0 hi (stride1=0)
  g1[7] = 0;
  i32x4 gz; gz[0] = 0; gz[1] = 0; gz[2] = 0; gz[3] = 0;
  i32x8 gz8;
  for (int i = 0; i < 8; ++i) gz8[i] = 0;
  __builtin_amdgcn_tensor_load_to_lds(g0, g1, gz, gz, gz8, 0);
}
#endif

// ---------------- pack / convert kernels ----------------
__global__ void k_f2bf(const float* __restrict__ src, unsigned short* __restrict__ dst, long n) {
  for (long i = (long)blockIdx.x * blockDim.x + threadIdx.x; i < n; i += (long)gridDim.x * blockDim.x)
    dst[i] = f2bf(src[i]);
}

__global__ void k_pack_h(const float* __restrict__ mem, const float* __restrict__ x,
                         unsigned short* __restrict__ h) {
  const long n = (long)BATCH * TOT * D_MODEL;
  for (long i = (long)blockIdx.x * blockDim.x + threadIdx.x; i < n; i += (long)gridDim.x * blockDim.x) {
    long b = i / ((long)TOT * D_MODEL);
    long rem = i - b * (long)TOT * D_MODEL;
    long j = rem >> 10;
    long c = rem & (D_MODEL - 1);
    float v = (j < PREV) ? mem[(b * PREV + j) * D_MODEL + c]
                         : x[(b * CUR + (j - PREV)) * D_MODEL + c];
    h[i] = f2bf(v);
  }
}

__global__ void k_relshift(const unsigned short* __restrict__ qv, unsigned short* __restrict__ qvsh) {
  const int row  = blockIdx.x;
  const int rsrc = BATCH + row;
  const int bb = rsrc / (CUR + 1);
  const int ii = rsrc % (CUR + 1);
  uint2* dst = reinterpret_cast<uint2*>(qvsh + (long)row * D_INNER);
  if (ii == 0) {
    uint2 z; z.x = 0u; z.y = 0u;
    dst[threadIdx.x] = z;
  } else {
    const uint2* src = reinterpret_cast<const uint2*>(qv + (long)(bb * CUR + ii - 1) * D_INNER);
    dst[threadIdx.x] = src[threadIdx.x];
  }
}

// ---------------- bf16 WMMA GEMM: C = A(MxK) * B^T (B stored NxK row-major) ----------------
// WG tile 128x64, 8 waves in 4x2 grid, each wave owns 32x32 (2x2 WMMA tiles).
// Double-buffered LDS tiles filled by async global->LDS DMA overlapped with WMMA.
__global__ __launch_bounds__(256)
void k_gemm_bf16(const unsigned short* __restrict__ A,
                 const unsigned short* __restrict__ Bm,
                 int M, int N, int K,
                 unsigned short* __restrict__ out1, const float* __restrict__ bias1,
                 unsigned short* __restrict__ out2, const float* __restrict__ bias2,
                 float* __restrict__ outf, const float* __restrict__ resid,
                 const float* __restrict__ fbias)
{
  __shared__ alignas(16) unsigned short As[2][128 * 32];
  __shared__ alignas(16) unsigned short Bs[2][64 * 32];

  const int nTiles = N >> 6;
  const int nb = blockIdx.x % nTiles;
  const int mb = blockIdx.x / nTiles;
  const int m0 = mb << 7, n0 = nb << 6;
  const int tid = threadIdx.x;
  const int lane = tid & 31;
  const int w = tid >> 5;
  const int wm = w & 3, wn = w >> 2;

  f32x8 acc[2][2];
  for (int i = 0; i < 2; ++i)
    for (int j = 0; j < 2; ++j)
      for (int r = 0; r < 8; ++r) acc[i][j][r] = 0.f;

  // per-thread async staging chunks (b128 = 8 bf16)
  const int c0 = tid * 2, c1 = tid * 2 + 1;
  const int aR0 = c0 >> 2, aC0 = (c0 & 3) << 3;
  const int aR1 = c1 >> 2, aC1 = (c1 & 3) << 3;
  const int bR = tid >> 2, bC = (tid & 3) << 3;

  const unsigned short* gA0 = A + (long)(m0 + aR0) * K + aC0;
  const unsigned short* gA1 = A + (long)(m0 + aR1) * K + aC1;
  const unsigned short* gB  = Bm + (long)(n0 + bR) * K + bC;

  const unsigned asBase = (unsigned)(uintptr_t)&As[0][0];
  const unsigned bsBase = (unsigned)(uintptr_t)&Bs[0][0];
  const unsigned aOff0 = (unsigned)(aR0 * 32 + aC0) * 2u;
  const unsigned aOff1 = (unsigned)(aR1 * 32 + aC1) * 2u;
  const unsigned bOff  = (unsigned)(bR * 32 + bC) * 2u;

  // prologue: stage k=0 into buffer 0
  async_load_b128(asBase + aOff0, gA0);
  async_load_b128(asBase + aOff1, gA1);
  async_load_b128(bsBase + bOff,  gB);

  int buf = 0;
  for (int kk = 0; kk < K; kk += 32) {
    if (kk + 32 < K) {  // stage next tile into the other buffer while computing
      const unsigned ao = (unsigned)((buf ^ 1) * 128 * 32 * 2);
      const unsigned bo = (unsigned)((buf ^ 1) * 64 * 32 * 2);
      async_load_b128(asBase + ao + aOff0, gA0 + kk + 32);
      async_load_b128(asBase + ao + aOff1, gA1 + kk + 32);
      async_load_b128(bsBase + bo + bOff,  gB  + kk + 32);
      asm volatile("s_wait_asynccnt 0x3" ::: "memory"); // current buffer resident
    } else {
      asm volatile("s_wait_asynccnt 0x0" ::: "memory");
    }
    __syncthreads();

    const unsigned short* Ab = &As[buf][0] + (wm * 32) * 32;
    const unsigned short* Bb = &Bs[buf][0] + (wn * 32) * 32;
    v16bf a0 = load_a_frag(Ab, 32);
    v16bf a1 = load_a_frag(Ab + 16 * 32, 32);
    v16bf b0 = load_b_frag(Bb, 32);
    v16bf b1 = load_b_frag(Bb + 16 * 32, 32);
    acc[0][0] = wmma_bf16(a0, b0, acc[0][0]);
    acc[0][1] = wmma_bf16(a0, b1, acc[0][1]);
    acc[1][0] = wmma_bf16(a1, b0, acc[1][0]);
    acc[1][1] = wmma_bf16(a1, b1, acc[1][1]);

    __syncthreads();   // all waves done reading before next DMA overwrites
    buf ^= 1;
  }

  for (int sm = 0; sm < 2; ++sm)
    for (int sn = 0; sn < 2; ++sn) {
      const int crow = m0 + wm * 32 + sm * 16 + ((lane >> 4) << 3);
      const int ccol = n0 + wn * 32 + sn * 16 + (lane & 15);
      for (int r = 0; r < 8; ++r) {
        const long idx = (long)(crow + r) * N + ccol;
        const float cv = acc[sm][sn][r];
        if (out1) out1[idx] = f2bf(cv + (bias1 ? bias1[ccol] : 0.f));
        if (out2) out2[idx] = f2bf(cv + (bias2 ? bias2[ccol] : 0.f));
        if (outf) outf[idx] = cv + (resid ? resid[idx] : 0.f) + (fbias ? fbias[ccol] : 0.f);
      }
    }
}

// stage 64x64 bf16 tiles of K and Pos into LDS (TDM when available)
__device__ __forceinline__ void stage_kp_tiles(unsigned short* Kt, unsigned short* Pt,
                                               const unsigned short* kv, const unsigned short* pos,
                                               int b, int h, int j0) {
  const int tid = threadIdx.x;
#if HAS_TDM
  if ((tid >> 5) == 0) {  // wave 0 drives the DMA engine (EXEC ignored by TDM)
    tdm_load_tile_2d((unsigned)(uintptr_t)Kt,
                     kv + (long)(b * TOT + j0) * (2 * D_INNER) + h * D_HEAD,
                     64u, 64u, (unsigned long long)(2 * D_INNER));
    tdm_load_tile_2d((unsigned)(uintptr_t)Pt,
                     pos + (long)j0 * D_INNER + h * D_HEAD,
                     64u, 64u, (unsigned long long)D_INNER);
    __builtin_amdgcn_s_wait_tensorcnt((short)0);
  }
#else
  const int jl = tid >> 2, dc = (tid & 3) << 4;
  const unsigned short* ksrc = kv + ((long)(b * TOT + j0 + jl) * (2 * D_INNER)) + h * D_HEAD + dc;
  *reinterpret_cast<uint4*>(Kt + jl * 64 + dc)     = *reinterpret_cast<const uint4*>(ksrc);
  *reinterpret_cast<uint4*>(Kt + jl * 64 + dc + 8) = *reinterpret_cast<const uint4*>(ksrc + 8);
  const unsigned short* psrc = pos + (long)(j0 + jl) * D_INNER + h * D_HEAD + dc;
  *reinterpret_cast<uint4*>(Pt + jl * 64 + dc)     = *reinterpret_cast<const uint4*>(psrc);
  *reinterpret_cast<uint4*>(Pt + jl * 64 + dc + 8) = *reinterpret_cast<const uint4*>(psrc + 8);
#endif
}

// ---------------- attention column-softmax stats (softmax over query axis i) ----------------
__global__ __launch_bounds__(256)
void k_attn_stats(const unsigned short* __restrict__ qu,
                  const unsigned short* __restrict__ qvsh,
                  const unsigned short* __restrict__ kv,
                  const unsigned short* __restrict__ pos,
                  float* __restrict__ colmax, float* __restrict__ colsum)
{
  __shared__ alignas(16) unsigned short Kt[64 * 64];
  __shared__ alignas(16) unsigned short Pt[64 * 64];
  __shared__ float red[8 * 64];
  __shared__ float cm[64];

  const int njt = TOT / 64;
  int bid = blockIdx.x;
  const int jt = bid % njt; bid /= njt;
  const int h  = bid % N_HEADS;
  const int b  = bid / N_HEADS;
  const int j0 = jt << 6;
  const int tid = threadIdx.x, lane = tid & 31, w = tid >> 5;
  const int colLane = lane & 15;

  stage_kp_tiles(Kt, Pt, kv, pos, b, h, j0);
  __syncthreads();

  // pass 1: per-column max over i
  float mloc[4]; for (int n = 0; n < 4; ++n) mloc[n] = -3.0e38f;
  for (int it = w; it < CUR / 16; it += 8) {
    const int i0 = it << 4;
    const unsigned short* quB = qu   + (long)(b * CUR + i0) * D_INNER + h * D_HEAD;
    const unsigned short* qvB = qvsh + (long)(b * CUR + i0) * D_INNER + h * D_HEAD;
    v16bf aq0 = load_a_frag(quB, D_INNER);
    v16bf aq1 = load_a_frag(quB + 32, D_INNER);
    v16bf av0 = load_a_frag(qvB, D_INNER);
    v16bf av1 = load_a_frag(qvB + 32, D_INNER);
    for (int nbI = 0; nbI < 4; ++nbI) {
      f32x8 acc; for (int r = 0; r < 8; ++r) acc[r] = 0.f;
      acc = wmma_bf16(aq0, load_b_frag(Kt + nbI * 16 * 64, 64), acc);
      acc = wmma_bf16(aq1, load_b_frag(Kt + nbI * 16 * 64 + 32, 64), acc);
      acc = wmma_bf16(av0, load_b_frag(Pt + nbI * 16 * 64, 64), acc);
      acc = wmma_bf16(av1, load_b_frag(Pt + nbI * 16 * 64 + 32, 64), acc);
      const int gj = j0 + nbI * 16 + colLane;
      float m = -3.0e38f;
      for (int r = 0; r < 8; ++r) {
        const int gi = i0 + ((lane >> 4) << 3) + r;
        float sv = (gj > gi + PREV) ? -3.0e38f : acc[r] * 0.125f;
        m = fmaxf(m, sv);
      }
      m = fmaxf(m, __shfl_xor(m, 16));
      mloc[nbI] = fmaxf(mloc[nbI], m);
    }
  }
  if (lane < 16)
    for (int nbI = 0; nbI < 4; ++nbI) red[w * 64 + nbI * 16 + colLane] = mloc[nbI];
  __syncthreads();
  if (tid < 64) {
    float m = -3.0e38f;
    for (int w2 = 0; w2 < 8; ++w2) m = fmaxf(m, red[w2 * 64 + tid]);
    cm[tid] = m;
  }
  __syncthreads();

  // pass 2: per-column sum of exp
  float sloc[4]; for (int n = 0; n < 4; ++n) sloc[n] = 0.f;
  for (int it = w; it < CUR / 16; it += 8) {
    const int i0 = it << 4;
    const unsigned short* quB = qu   + (long)(b * CUR + i0) * D_INNER + h * D_HEAD;
    const unsigned short* qvB = qvsh + (long)(b * CUR + i0) * D_INNER + h * D_HEAD;
    v16bf aq0 = load_a_frag(quB, D_INNER);
    v16bf aq1 = load_a_frag(quB + 32, D_INNER);
    v16bf av0 = load_a_frag(qvB, D_INNER);
    v16bf av1 = load_a_frag(qvB + 32, D_INNER);
    for (int nbI = 0; nbI < 4; ++nbI) {
      f32x8 acc; for (int r = 0; r < 8; ++r) acc[r] = 0.f;
      acc = wmma_bf16(aq0, load_b_frag(Kt + nbI * 16 * 64, 64), acc);
      acc = wmma_bf16(aq1, load_b_frag(Kt + nbI * 16 * 64 + 32, 64), acc);
      acc = wmma_bf16(av0, load_b_frag(Pt + nbI * 16 * 64, 64), acc);
      acc = wmma_bf16(av1, load_b_frag(Pt + nbI * 16 * 64 + 32, 64), acc);
      const int gj = j0 + nbI * 16 + colLane;
      const float cmv = cm[nbI * 16 + colLane];
      float s = 0.f;
      for (int r = 0; r < 8; ++r) {
        const int gi = i0 + ((lane >> 4) << 3) + r;
        float sv = (gj > gi + PREV) ? -3.0e38f : acc[r] * 0.125f;
        s += __expf(sv - cmv);
      }
      s += __shfl_xor(s, 16);
      sloc[nbI] += s;
    }
  }
  if (lane < 16)
    for (int nbI = 0; nbI < 4; ++nbI) red[w * 64 + nbI * 16 + colLane] = sloc[nbI];
  __syncthreads();
  if (tid < 64) {
    float s = 0.f;
    for (int w2 = 0; w2 < 8; ++w2) s += red[w2 * 64 + tid];
    const long o = ((long)(b * N_HEADS + h)) * TOT + j0 + tid;
    colmax[o] = cm[tid];
    colsum[o] = s;
  }
}

// ---------------- attention output: O = exp(S - m_j) * (V / sum_j) ----------------
__global__ __launch_bounds__(256)
void k_attn_out(const unsigned short* __restrict__ qu,
                const unsigned short* __restrict__ qvsh,
                const unsigned short* __restrict__ kv,
                const unsigned short* __restrict__ pos,
                const float* __restrict__ colmax, const float* __restrict__ colsum,
                unsigned short* __restrict__ attno)
{
  __shared__ alignas(16) unsigned short Kt[64 * 64];
  __shared__ alignas(16) unsigned short Pt[64 * 64];
  __shared__ alignas(16) unsigned short Vt[64 * 64];
  __shared__ alignas(16) unsigned short Et[8 * 16 * 64];
  __shared__ float cm[64], csr[64];

  const int nit = CUR / 64;
  int bid = blockIdx.x;
  const int itb = bid % nit; bid /= nit;
  const int h = bid % N_HEADS;
  const int b = bid / N_HEADS;
  const int i0 = itb << 6;
  const int tid = threadIdx.x, lane = tid & 31, w = tid >> 5;
  const int rb = w >> 1;
  const int dh = w & 1;
  const int colLane = lane & 15;

  f32x8 accd0, accd1;
  for (int r = 0; r < 8; ++r) { accd0[r] = 0.f; accd1[r] = 0.f; }

  const int iRow = i0 + rb * 16;
  const unsigned short* quB = qu   + (long)(b * CUR + iRow) * D_INNER + h * D_HEAD;
  const unsigned short* qvB = qvsh + (long)(b * CUR + iRow) * D_INNER + h * D_HEAD;
  v16bf aq0 = load_a_frag(quB, D_INNER);
  v16bf aq1 = load_a_frag(quB + 32, D_INNER);
  v16bf av0 = load_a_frag(qvB, D_INNER);
  v16bf av1 = load_a_frag(qvB + 32, D_INNER);

  for (int jt = 0; jt < TOT / 64; ++jt) {
    const int j0 = jt << 6;
    __syncthreads();
    stage_kp_tiles(Kt, Pt, kv, pos, b, h, j0);
    if (tid < 64) {
      const long o = ((long)(b * N_HEADS + h)) * TOT + j0 + tid;
      cm[tid]  = colmax[o];
      csr[tid] = 1.0f / colsum[o];
    }
    __syncthreads();
    // stage V transposed [d][j] with 1/colsum folded in
    for (int idx = tid; idx < 64 * 64; idx += 256) {
      const int d = idx >> 6, jl = idx & 63;
      const float vv = bf2f(kv[(long)(b * TOT + j0 + jl) * (2 * D_INNER) + D_INNER + h * D_HEAD + d]);
      Vt[d * 64 + jl] = f2bf(vv * csr[jl]);
    }
    __syncthreads();

    for (int nbI = 0; nbI < 4; ++nbI) {
      f32x8 acc; for (int r = 0; r < 8; ++r) acc[r] = 0.f;
      acc = wmma_bf16(aq0, load_b_frag(Kt + nbI * 16 * 64, 64), acc);
      acc = wmma_bf16(aq1, load_b_frag(Kt + nbI * 16 * 64 + 32, 64), acc);
      acc = wmma_bf16(av0, load_b_frag(Pt + nbI * 16 * 64, 64), acc);
      acc = wmma_bf16(av1, load_b_frag(Pt + nbI * 16 * 64 + 32, 64), acc);
      const int gj = j0 + nbI * 16 + colLane;
      const float cmv = cm[nbI * 16 + colLane];
      for (int r = 0; r < 8; ++r) {
        const int erow = ((lane >> 4) << 3) + r;
        const int gi = iRow + erow;
        float sv = (gj > gi + PREV) ? -3.0e38f : acc[r] * 0.125f;
        Et[w * 1024 + erow * 64 + nbI * 16 + colLane] = f2bf(__expf(sv - cmv));
      }
    }
    asm volatile("s_wait_dscnt 0x0" ::: "memory");

    for (int kk = 0; kk < 64; kk += 32) {
      v16bf ea = load_a_frag(Et + w * 1024 + kk, 64);
      accd0 = wmma_bf16(ea, load_b_frag(Vt + (dh * 32 + 0)  * 64 + kk, 64), accd0);
      accd1 = wmma_bf16(ea, load_b_frag(Vt + (dh * 32 + 16) * 64 + kk, 64), accd1);
    }
  }

  const int crow = iRow + ((lane >> 4) << 3);
  for (int r = 0; r < 8; ++r) {
    const long row = (long)(b * CUR + crow + r);
    attno[row * D_INNER + h * D_HEAD + dh * 32 + colLane]      = f2bf(accd0[r]);
    attno[row * D_INNER + h * D_HEAD + dh * 32 + 16 + colLane] = f2bf(accd1[r]);
  }
}

// ---------------- LayerNorm over last dim (1024) ----------------
__global__ __launch_bounds__(256)
void k_ln(const float* __restrict__ y, const float* __restrict__ gamma,
          const float* __restrict__ beta, float* __restrict__ out)
{
  __shared__ float red[256];
  const int row = blockIdx.x;
  const int t = threadIdx.x;
  const float* yr = y + (long)row * D_MODEL;
  float4 vv = *reinterpret_cast<const float4*>(yr + t * 4);
  red[t] = vv.x + vv.y + vv.z + vv.w;
  __syncthreads();
  for (int off = 128; off > 0; off >>= 1) {
    if (t < off) red[t] += red[t + off];
    __syncthreads();
  }
  const float mu = red[0] * (1.0f / D_MODEL);
  __syncthreads();
  float d0 = vv.x - mu, d1 = vv.y - mu, d2 = vv.z - mu, d3 = vv.w - mu;
  red[t] = d0 * d0 + d1 * d1 + d2 * d2 + d3 * d3;
  __syncthreads();
  for (int off = 128; off > 0; off >>= 1) {
    if (t < off) red[t] += red[t + off];
    __syncthreads();
  }
  const float rstd = rsqrtf(red[0] * (1.0f / D_MODEL) + 1e-5f);
  float4 g  = *reinterpret_cast<const float4*>(gamma + t * 4);
  float4 bt = *reinterpret_cast<const float4*>(beta + t * 4);
  float4 o;
  o.x = d0 * rstd * g.x + bt.x;
  o.y = d1 * rstd * g.y + bt.y;
  o.z = d2 * rstd * g.z + bt.z;
  o.w = d3 * rstd * g.w + bt.w;
  *reinterpret_cast<float4*>(out + (long)row * D_MODEL + t * 4) = o;
}

// ---------------- host launcher ----------------
extern "C" void kernel_launch(void* const* d_in, const int* in_sizes, int n_in,
                              void* d_out, int out_size, void* d_ws, size_t ws_size,
                              hipStream_t stream) {
  (void)in_sizes; (void)n_in; (void)out_size; (void)ws_size;
  const float* x     = (const float*)d_in[0];
  const float* pos   = (const float*)d_in[1];
  const float* u     = (const float*)d_in[2];
  const float* v     = (const float*)d_in[3];
  const float* mem   = (const float*)d_in[4];
  /* d_in[5] = tgt_mask (computed analytically) */
  const float* Wq    = (const float*)d_in[6];
  const float* Wkv   = (const float*)d_in[7];
  const float* Wfc   = (const float*)d_in[8];
  const float* bfc   = (const float*)d_in[9];
  const float* gamma = (const float*)d_in[10];
  const float* beta  = (const float*)d_in[11];
  float* out = (float*)d_out;

  char* p = (char*)d_ws;
  auto alloc = [&](size_t bytes) -> char* {
    char* r = p; p += (bytes + 255) & ~(size_t)255; return r;
  };
  unsigned short* xbf   = (unsigned short*)alloc((size_t)BATCH * CUR * D_MODEL * 2);
  unsigned short* hbf   = (unsigned short*)alloc((size_t)BATCH * TOT * D_MODEL * 2);
  unsigned short* wqbf  = (unsigned short*)alloc((size_t)D_INNER * D_MODEL * 2);
  unsigned short* wkvbf = (unsigned short*)alloc((size_t)2 * D_INNER * D_MODEL * 2);
  unsigned short* wfcbf = (unsigned short*)alloc((size_t)D_MODEL * D_INNER * 2);
  unsigned short* posbf = (unsigned short*)alloc((size_t)TOT * D_INNER * 2);
  unsigned short* qu    = (unsigned short*)alloc((size_t)BATCH * CUR * D_INNER * 2);
  unsigned short* qv    = (unsigned short*)alloc((size_t)BATCH * CUR * D_INNER * 2);
  unsigned short* qvsh  = (unsigned short*)alloc((size_t)BATCH * CUR * D_INNER * 2);
  unsigned short* kvbf  = (unsigned short*)alloc((size_t)BATCH * TOT * 2 * D_INNER * 2);
  unsigned short* attno = (unsigned short*)alloc((size_t)BATCH * CUR * D_INNER * 2);
  float* colmax = (float*)alloc((size_t)BATCH * N_HEADS * TOT * 4);
  float* colsum = (float*)alloc((size_t)BATCH * N_HEADS * TOT * 4);
  float* yb     = (float*)alloc((size_t)BATCH * CUR * D_MODEL * 4);

  k_f2bf<<<1024, 256, 0, stream>>>(x,   xbf,   (long)BATCH * CUR * D_MODEL);
  k_f2bf<<<1024, 256, 0, stream>>>(Wq,  wqbf,  (long)D_INNER * D_MODEL);
  k_f2bf<<<1024, 256, 0, stream>>>(Wkv, wkvbf, (long)2 * D_INNER * D_MODEL);
  k_f2bf<<<1024, 256, 0, stream>>>(Wfc, wfcbf, (long)D_MODEL * D_INNER);
  k_f2bf<<<1024, 256, 0, stream>>>(pos, posbf, (long)TOT * D_INNER);
  k_pack_h<<<2048, 256, 0, stream>>>(mem, x, hbf);

  // q projections with u/v biases fused -> qu, qv (bf16)
  k_gemm_bf16<<<(D_INNER / 64) * ((BATCH * CUR) / 128), 256, 0, stream>>>(
      xbf, wqbf, BATCH * CUR, D_INNER, D_MODEL,
      qu, u, qv, v, nullptr, nullptr, nullptr);
  // kv projection
  k_gemm_bf16<<<((2 * D_INNER) / 64) * ((BATCH * TOT) / 128), 256, 0, stream>>>(
      hbf, wkvbf, BATCH * TOT, 2 * D_INNER, D_MODEL,
      kvbf, nullptr, nullptr, nullptr, nullptr, nullptr, nullptr);

  k_relshift<<<BATCH * CUR, 256, 0, stream>>>(qv, qvsh);

  k_attn_stats<<<BATCH * N_HEADS * (TOT / 64), 256, 0, stream>>>(
      qu, qvsh, kvbf, posbf, colmax, colsum);
  k_attn_out<<<BATCH * N_HEADS * (CUR / 64), 256, 0, stream>>>(
      qu, qvsh, kvbf, posbf, colmax, colsum, attno);

  // FC projection + residual + bias (fp32 out)
  k_gemm_bf16<<<(D_MODEL / 64) * ((BATCH * CUR) / 128), 256, 0, stream>>>(
      attno, wfcbf, BATCH * CUR, D_MODEL, D_INNER,
      nullptr, nullptr, nullptr, nullptr, yb, x, bfc);

  k_ln<<<BATCH * CUR, 256, 0, stream>>>(yb, gamma, beta, out);
}
